// AttentionLayer_42116449305335
// MI455X (gfx1250) — matrix operand order repaired and checked by reading, not compiled, not run on hardware
//
#include <hip/hip_runtime.h>
#include <hip/hip_bf16.h>

typedef __attribute__((ext_vector_type(16))) __bf16 v16bf;
typedef __attribute__((ext_vector_type(8)))  __bf16 v8bf;
typedef __attribute__((ext_vector_type(8)))  float  v8f;

#define HEADS   32
#define HIDDEN  3072
#define HEAD_DIM 96
#define HALF_DIM 48
#define BATCH   2
#define SEQ     2048
#define MTOT    (BATCH*SEQ)          // 4096 rows
#define QK_ELEMS ((size_t)BATCH*HEADS*SEQ*HEAD_DIM)   // 12,582,912
#define QK_BYTES (QK_ELEMS*2)                          // 25,165,824

union BF16x16 { v16bf v; v8bf h[2]; };

static __device__ __forceinline__ v16bf load_pair(const __bf16* p0, const __bf16* p1) {
  BF16x16 u;
  u.h[0] = *reinterpret_cast<const v8bf*>(p0);
  u.h[1] = *reinterpret_cast<const v8bf*>(p1);
  return u.v;
}

static __device__ __forceinline__ v8f wmma_bf16(v16bf a, v16bf b, v8f c) {
  return __builtin_amdgcn_wmma_f32_16x16x32_bf16(false, a, false, b, (short)0, c, false, false);
}

// ---------------------------------------------------------------------------
// Kernel 1: Q/K projection + fused RoPE, register-blocked 2 M-tiles per wave.
// One wave computes rows [m0, m0+32) of the RoPE partner pair of columns
// (cols d, d+48 of one head) of C = x @ w_qkv^T.  Writes:
//   qro [b,h,s,d]  roped Q, pre-scaled by 1/sqrt(96)
//   kro [b,h,s,d]  roped K               (B-fragments of Q.K^T)
//   kT  [b,h,d,s]  UN-roped K transposed (B-fragments of P.V; v = pre-rope k)
// ---------------------------------------------------------------------------
__global__ __launch_bounds__(256)
void qk_rope_kernel(const __bf16* __restrict__ x, const __bf16* __restrict__ w_qkv,
                    __bf16* __restrict__ qro, __bf16* __restrict__ kro,
                    __bf16* __restrict__ kT)
{
  const int NPAIR = 2 * HEADS * 3;                         // 192 col-pair tiles
  const int lane  = threadIdx.x & 31;
  const int wave  = blockIdx.x * (blockDim.x >> 5) + (threadIdx.x >> 5);

  const int mt  = wave / NPAIR;          // 0..127  (32-row block of 4096)
  const int ct  = wave % NPAIR;
  const int mat = ct / 96;               // 0 = Q, 1 = K
  const int h   = (ct % 96) / 3;
  const int p   = (ct % 96) % 3;         // 16-col pair within the 48-half

  const int m0  = mt * 32;
  const int nA  = mat * HIDDEN + h * HEAD_DIM + p * 16;    // w_qkv row block A
  const int nB  = nA + HALF_DIM;                           // RoPE partner block

  const int l16 = lane & 15;
  const int hi  = lane >> 4;
  const int aoff = hi * 8;               // A frag K offsets per ISA layout
  const int boff = hi * 8;               // B frag K offsets per ISA layout

  const __bf16* arow0 = x     + (size_t)(m0 + l16) * HIDDEN;
  const __bf16* arow1 = x     + (size_t)(m0 + 16 + l16) * HIDDEN;
  const __bf16* brA   = w_qkv + (size_t)(nA + l16) * HIDDEN;
  const __bf16* brB   = w_qkv + (size_t)(nB + l16) * HIDDEN;

  v8f accA0 = {}, accB0 = {}, accA1 = {}, accB1 = {};
#pragma unroll 4
  for (int k0 = 0; k0 < HIDDEN; k0 += 32) {
    v16bf af0 = load_pair(arow0 + k0 + aoff, arow0 + k0 + 16 + aoff);
    v16bf af1 = load_pair(arow1 + k0 + aoff, arow1 + k0 + 16 + aoff);
    v16bf bA  = load_pair(brA + k0 + boff, brA + k0 + boff + 16);
    v16bf bB  = load_pair(brB + k0 + boff, brB + k0 + boff + 16);
    accA0 = wmma_bf16(af0, bA, accA0);
    accB0 = wmma_bf16(af0, bB, accB0);
    accA1 = wmma_bf16(af1, bA, accA1);
    accB1 = wmma_bf16(af1, bB, accB1);
  }

  // RoPE epilogue. freq index = d1 = p*16 + l16 in [0,48)
  const int d1 = p * 16 + l16;
  const float inv_freq = __expf(-(float)d1 * 0.19188209108283716f); // ln(1e4)/48
  const float qsc = (mat == 0) ? 0.10206207261596575f : 1.0f;       // 1/sqrt(96)
  __bf16* dst = (mat == 0) ? qro : kro;

#pragma unroll
  for (int t = 0; t < 2; ++t) {
    const v8f& aA = t ? accA1 : accA0;
    const v8f& aB = t ? accB1 : accB0;
#pragma unroll
    for (int r = 0; r < 8; ++r) {
      const int m = m0 + t * 16 + r + hi * 8;
      const int b = m >> 11;
      const int s = m & (SEQ - 1);
      float sn, cs;
      __sincosf((float)s * inv_freq, &sn, &cs);
      const float t1 = aA[r], t2 = aB[r];
      const float o1 = (t1 * cs - t2 * sn) * qsc;
      const float o2 = (t2 * cs + t1 * sn) * qsc;
      const size_t base = (((size_t)b * HEADS + h) * SEQ + s) * HEAD_DIM;
      dst[base + d1]            = (__bf16)o1;
      dst[base + d1 + HALF_DIM] = (__bf16)o2;
      if (mat == 1) {  // V = pre-RoPE K, stored transposed [b,h,d,s]
        const size_t tb = ((size_t)b * HEADS + h) * HEAD_DIM;
        kT[(tb + d1)            * SEQ + s] = (__bf16)t1;
        kT[(tb + d1 + HALF_DIM) * SEQ + s] = (__bf16)t2;
      }
    }
  }
}

// ---------------------------------------------------------------------------
// Kernel 2: causal flash attention. One wave per (b, h, 16-row Q tile).
// Keys processed in chunks of 32 (two 16x16 score tiles -> one 16x32 P A-frag).
// ---------------------------------------------------------------------------
__global__ __launch_bounds__(32)
void flash_attn_kernel(const __bf16* __restrict__ qro, const __bf16* __restrict__ kro,
                       const __bf16* __restrict__ kT, __bf16* __restrict__ att)
{
  __shared__ __bf16 pls[16 * 32];        // P tile bounce: C-layout -> A-layout

  const int lane = threadIdx.x & 31;
  const int l16  = lane & 15;
  const int hi   = lane >> 4;
  int bid = blockIdx.x;
  const int qt = bid & 127;  bid >>= 7;
  const int h  = bid & 31;   bid >>= 5;
  const int b  = bid;
  const int q0 = qt * 16;

  const __bf16* qbase  = qro + ((size_t)b * HEADS + h) * SEQ * HEAD_DIM;
  const __bf16* kbase  = kro + ((size_t)b * HEADS + h) * SEQ * HEAD_DIM;
  const __bf16* vTbase = kT  + ((size_t)b * HEADS + h) * HEAD_DIM * SEQ;

  const int aoff = hi * 8;
  const int boff = hi * 8;

  // Q fragments for the 96-wide K-dim (3 x k=32), kept in registers.
  v16bf qf[3];
  {
    const __bf16* qrow = qbase + (size_t)(q0 + l16) * HEAD_DIM;
#pragma unroll
    for (int i = 0; i < 3; ++i)
      qf[i] = load_pair(qrow + i * 32 + aoff, qrow + i * 32 + 16 + aoff);
  }

  v8f o[6];
#pragma unroll
  for (int d = 0; d < 6; ++d) o[d] = (v8f){};
  float mrow[8], lrow[8];
#pragma unroll
  for (int r = 0; r < 8; ++r) { mrow[r] = -3.0e38f; lrow[r] = 0.0f; }

  const int nchunks = q0 / 32 + 1;       // causal: keys <= q0+15
  for (int c = 0; c < nchunks; ++c) {
    const int k0 = c * 32;

    // S = Q . K^T  (two key tiles)
    v8f s0 = (v8f){}, s1 = (v8f){};
    const __bf16* kr0 = kbase + (size_t)(k0 + l16) * HEAD_DIM;
    const __bf16* kr1 = kbase + (size_t)(k0 + 16 + l16) * HEAD_DIM;
#pragma unroll
    for (int i = 0; i < 3; ++i) {
      v16bf b0 = load_pair(kr0 + i * 32 + boff, kr0 + i * 32 + boff + 16);
      v16bf b1 = load_pair(kr1 + i * 32 + boff, kr1 + i * 32 + boff + 16);
      s0 = wmma_bf16(qf[i], b0, s0);
      s1 = wmma_bf16(qf[i], b1, s1);
    }

    // online softmax (row reductions across the 16 lanes of each half-wave)
    float sc[8];
    const int key0 = k0 + l16;
    const int key1 = k0 + 16 + l16;
#pragma unroll
    for (int r = 0; r < 8; ++r) {
      const int row = q0 + r + hi * 8;
      float v0 = (key0 <= row) ? s0[r] : -3.0e38f;
      float v1 = (key1 <= row) ? s1[r] : -3.0e38f;
      float mx = fmaxf(v0, v1);
#pragma unroll
      for (int off = 1; off < 16; off <<= 1)
        mx = fmaxf(mx, __shfl_xor(mx, off, 32));
      const float mnew = fmaxf(mrow[r], mx);
      sc[r] = __expf(mrow[r] - mnew);
      mrow[r] = mnew;
      const float p0 = (key0 <= row) ? __expf(s0[r] - mnew) : 0.0f;
      const float p1 = (key1 <= row) ? __expf(s1[r] - mnew) : 0.0f;
      float ps = p0 + p1;
#pragma unroll
      for (int off = 1; off < 16; off <<= 1)
        ps += __shfl_xor(ps, off, 32);
      lrow[r] = lrow[r] * sc[r] + ps;
      pls[(r + hi * 8) * 32 + l16]      = (__bf16)p0;
      pls[(r + hi * 8) * 32 + 16 + l16] = (__bf16)p1;
    }
#pragma unroll
    for (int d = 0; d < 6; ++d)
#pragma unroll
      for (int r = 0; r < 8; ++r) o[d][r] *= sc[r];

    __syncthreads();   // single-wave WG: forces DS waits; barrier itself is a NOP

    // P A-fragment (16 rows x 32 keys) from LDS in ISA A-layout
    v16bf pf;
    {
      const __bf16* pr = &pls[(size_t)l16 * 32 + aoff];
      pf = load_pair(pr, pr + 16);
    }
    // O += P . V   (V = un-roped K, dim-major -> contiguous B-fragments)
#pragma unroll
    for (int d = 0; d < 6; ++d) {
      const __bf16* vr = vTbase + (size_t)(d * 16 + l16) * SEQ + k0 + boff;
      v16bf bf = load_pair(vr, vr + 16);
      o[d] = wmma_bf16(pf, bf, o[d]);
    }
    __syncthreads();   // protect pls before next chunk overwrites it
  }

  // normalize + store attn output in [b, s, h*96+d] layout (ready for w_o GEMM)
#pragma unroll
  for (int d = 0; d < 6; ++d) {
#pragma unroll
    for (int r = 0; r < 8; ++r) {
      const int row = q0 + r + hi * 8;
      const float v = o[d][r] / lrow[r];
      att[((size_t)b * SEQ + row) * HIDDEN + h * HEAD_DIM + d * 16 + l16] = (__bf16)v;
    }
  }
}

// ---------------------------------------------------------------------------
// Kernel 3: out = att @ w_o^T, register-blocked 2x2 tiles (32x32) per wave.
// ---------------------------------------------------------------------------
__global__ __launch_bounds__(256)
void out_proj_kernel(const __bf16* __restrict__ att, const __bf16* __restrict__ w_o,
                     __bf16* __restrict__ out)
{
  const int NT2  = HIDDEN / 32;                            // 96 col blocks
  const int lane = threadIdx.x & 31;
  const int wave = blockIdx.x * (blockDim.x >> 5) + (threadIdx.x >> 5);
  const int mt = wave / NT2;                               // 0..127
  const int nt = wave % NT2;                               // 0..95
  const int l16 = lane & 15;
  const int hi  = lane >> 4;
  const int aoff = hi * 8;
  const int boff = hi * 8;

  const __bf16* arow0 = att + (size_t)(mt * 32 + l16) * HIDDEN;
  const __bf16* arow1 = att + (size_t)(mt * 32 + 16 + l16) * HIDDEN;
  const __bf16* brow0 = w_o + (size_t)(nt * 32 + l16) * HIDDEN;
  const __bf16* brow1 = w_o + (size_t)(nt * 32 + 16 + l16) * HIDDEN;

  v8f acc00 = {}, acc01 = {}, acc10 = {}, acc11 = {};
#pragma unroll 4
  for (int k0 = 0; k0 < HIDDEN; k0 += 32) {
    v16bf af0 = load_pair(arow0 + k0 + aoff, arow0 + k0 + 16 + aoff);
    v16bf af1 = load_pair(arow1 + k0 + aoff, arow1 + k0 + 16 + aoff);
    v16bf bf0 = load_pair(brow0 + k0 + boff, brow0 + k0 + boff + 16);
    v16bf bf1 = load_pair(brow1 + k0 + boff, brow1 + k0 + boff + 16);
    acc00 = wmma_bf16(af0, bf0, acc00);
    acc01 = wmma_bf16(af0, bf1, acc01);
    acc10 = wmma_bf16(af1, bf0, acc10);
    acc11 = wmma_bf16(af1, bf1, acc11);
  }
#pragma unroll
  for (int r = 0; r < 8; ++r) {
    const size_t r0 = (size_t)(mt * 32 + r + hi * 8) * HIDDEN;
    const size_t r1 = (size_t)(mt * 32 + 16 + r + hi * 8) * HIDDEN;
    out[r0 + nt * 32 + l16]      = (__bf16)acc00[r];
    out[r0 + nt * 32 + 16 + l16] = (__bf16)acc01[r];
    out[r1 + nt * 32 + l16]      = (__bf16)acc10[r];
    out[r1 + nt * 32 + 16 + l16] = (__bf16)acc11[r];
  }
}

// ---------------------------------------------------------------------------
extern "C" void kernel_launch(void* const* d_in, const int* in_sizes, int n_in,
                              void* d_out, int out_size, void* d_ws, size_t ws_size,
                              hipStream_t stream)
{
  (void)in_sizes; (void)n_in; (void)out_size; (void)ws_size;
  const __bf16* x     = (const __bf16*)d_in[0];
  const __bf16* w_qkv = (const __bf16*)d_in[1];
  const __bf16* w_o   = (const __bf16*)d_in[2];
  __bf16* out = (__bf16*)d_out;

  char* ws = (char*)d_ws;
  __bf16* qro = (__bf16*)(ws);                  // roped Q   [b,h,s,d]
  __bf16* kro = (__bf16*)(ws + QK_BYTES);       // roped K   [b,h,s,d]
  __bf16* kT  = (__bf16*)(ws + 2 * QK_BYTES);   // raw   K^T [b,h,d,s]
  __bf16* att = (__bf16*)(ws + 3 * QK_BYTES);   // attn out  [b,s,h*d]

  // 128 32-row blocks * 192 col-pair tiles = 24576 waves, 8 waves/block
  qk_rope_kernel<<<3072, 256, 0, stream>>>(x, w_qkv, qro, kro, kT);
  // one wave per (b, h, q-tile): 2*32*128 = 8192
  flash_attn_kernel<<<8192, 32, 0, stream>>>(qro, kro, kT, att);
  // 128 x 96 blocks of 32x32 = 12288 waves, 8 waves/block
  out_proj_kernel<<<1536, 256, 0, stream>>>(att, w_o, out);
}
